// SelfAttention_29429115912623
// MI455X (gfx1250) — matrix operand order repaired and checked by reading, not compiled, not run on hardware
//
#include <hip/hip_runtime.h>

// ---------------------------------------------------------------------------
// Self-attention, MI455X (gfx1250, wave32, WMMA bf16 16x16x32).
// Phase 1: fused QKV projection GEMM  (f32 in -> bf16 Q,K,V^T in workspace)
//          with LDS-staged coalesced b128 epilogue stores.
// Phase 2: flash-attention (QK^T softmax PV fused, f32 out), async Q staging.
// ---------------------------------------------------------------------------

typedef __bf16 v16bf __attribute__((ext_vector_type(16)));
typedef __bf16 v8bf  __attribute__((ext_vector_type(8)));
typedef __bf16 v4bf  __attribute__((ext_vector_type(4)));
typedef float  v8f   __attribute__((ext_vector_type(8)));
typedef int    v4i_  __attribute__((vector_size(16)));   // matches builtin param

#define WMMA_BF16(a, b, c) \
    __builtin_amdgcn_wmma_f32_16x16x32_bf16(false, (a), false, (b), (short)0, (c), false, false)

#if __has_builtin(__builtin_amdgcn_global_load_async_to_lds_b128) && \
    __has_builtin(__builtin_amdgcn_s_wait_asynccnt)
#define HAVE_ASYNC_LDS 1
#else
#define HAVE_ASYNC_LDS 0
#endif

__device__ __forceinline__ v8f vzero8f() {
    v8f v = {0.f, 0.f, 0.f, 0.f, 0.f, 0.f, 0.f, 0.f};
    return v;
}

// A-matrix fragment (16x32 bf16, MxK). p = row base (row includes lane%16).
// Lane<16: K {0-7,16-23}; lane>=16: K {8-15,24-31}.  (ISA 7.12.2)
__device__ __forceinline__ v16bf fragA_ld(const __bf16* p, int lane) {
    const int half = (lane >> 4) & 1;
    v8bf lo = *(const v8bf*)(p + half * 8);
    v8bf hi = *(const v8bf*)(p + 16 + half * 8);
    return __builtin_shufflevector(lo, hi, 0,1,2,3,4,5,6,7,8,9,10,11,12,13,14,15);
}

// B-matrix fragment (32x16 bf16, KxN), column-major source: p = column base.
// Lane<16: K {0-7,16-23}; lane>=16: K {8-15,24-31}.
__device__ __forceinline__ v16bf fragB_ld(const __bf16* p, int lane) {
    const int half = (lane >> 4) & 1;
    v8bf lo = *(const v8bf*)(p + half * 8);
    v8bf hi = *(const v8bf*)(p + 16 + half * 8);
    return __builtin_shufflevector(lo, hi, 0,1,2,3,4,5,6,7,8,9,10,11,12,13,14,15);
}

// ---------------------------------------------------------------------------
// Phase 1: C[8192,1024] = X * W{q,k,v} + b.  Block tile 128x128, BK=32.
// 8 waves in a 4x2 grid, each wave 32x64 (2x4 WMMA tiles).
// z = 0 -> Q (row major bf16), 1 -> K (row major), 2 -> V transposed [B,D,S].
// ---------------------------------------------------------------------------
#define LDS_STRIDE 40   // 32 K + pad: 80B rows -> 16B aligned, conflict-free b128
#define CSTR 136        // C-tile stride (272B rows, 16B aligned)

__global__ __launch_bounds__(256) void qkv_gemm_kernel(
    const float* __restrict__ X,
    const float* __restrict__ Wq, const float* __restrict__ Wk, const float* __restrict__ Wv,
    const float* __restrict__ bq, const float* __restrict__ bk, const float* __restrict__ bv,
    __bf16* __restrict__ Qb, __bf16* __restrict__ Kb, __bf16* __restrict__ Vt)
{
    constexpr int D = 1024, S = 2048;
    const int m0 = blockIdx.x * 128;
    const int n0 = blockIdx.y * 128;
    const int z  = blockIdx.z;
    const float* W    = (z == 0) ? Wq : (z == 1) ? Wk : Wv;
    const float* bias = (z == 0) ? bq : (z == 1) ? bk : bv;

    // 34816B: lA(10240) + lB(10240) during the k-loop; reused as C tile after.
    __shared__ __align__(16) char p1smem[128 * CSTR * 2];
    __bf16* lA = (__bf16*)p1smem;
    __bf16* lB = (__bf16*)(p1smem + 128 * LDS_STRIDE * 2);

    const int tid  = threadIdx.x;
    const int lane = tid & 31;
    const int w    = tid >> 5;
    const int wm   = (w & 3) * 32;   // wave M offset within 128 tile
    const int wn   = (w >> 2) * 64;  // wave N offset within 128 tile

    v8f acc[2][4];
#pragma unroll
    for (int mt = 0; mt < 2; ++mt)
#pragma unroll
        for (int nt = 0; nt < 4; ++nt) acc[mt][nt] = vzero8f();

    const int aq = tid & 7;    // A: k-quad (8 x float4 = 32 K)
    const int ar = tid >> 3;   // A: row base 0..31
    const int bq4 = tid & 31;  // B: n-quad (32 x float4 = 128 N)
    const int bk8 = tid >> 5;  // B: k base 0..7

    for (int kt = 0; kt < D / 32; ++kt) {
        const int kbase = kt * 32;
#pragma unroll
        for (int rr = 0; rr < 4; ++rr) {
            const int row = ar + rr * 32;
            const float4 v = *(const float4*)(X + (size_t)(m0 + row) * D + kbase + aq * 4);
            v4bf a4;
            a4[0] = (__bf16)v.x; a4[1] = (__bf16)v.y;
            a4[2] = (__bf16)v.z; a4[3] = (__bf16)v.w;
            *(v4bf*)(&lA[row * LDS_STRIDE + aq * 4]) = a4;   // one ds_store_b64
        }
#pragma unroll
        for (int kk = 0; kk < 4; ++kk) {
            const int k = bk8 + kk * 8;
            const float4 v = *(const float4*)(W + (size_t)(kbase + k) * D + n0 + bq4 * 4);
            lB[(bq4 * 4 + 0) * LDS_STRIDE + k] = (__bf16)v.x;  // transpose scatter
            lB[(bq4 * 4 + 1) * LDS_STRIDE + k] = (__bf16)v.y;
            lB[(bq4 * 4 + 2) * LDS_STRIDE + k] = (__bf16)v.z;
            lB[(bq4 * 4 + 3) * LDS_STRIDE + k] = (__bf16)v.w;
        }
        __syncthreads();

        v16bf afrag[2], bfrag[4];
#pragma unroll
        for (int mt = 0; mt < 2; ++mt)
            afrag[mt] = fragA_ld(&lA[(wm + mt * 16 + (lane & 15)) * LDS_STRIDE], lane);
#pragma unroll
        for (int nt = 0; nt < 4; ++nt)
            bfrag[nt] = fragB_ld(&lB[(wn + nt * 16 + (lane & 15)) * LDS_STRIDE], lane);
#pragma unroll
        for (int mt = 0; mt < 2; ++mt)
#pragma unroll
            for (int nt = 0; nt < 4; ++nt)
                acc[mt][nt] = WMMA_BF16(afrag[mt], bfrag[nt], acc[mt][nt]);
        __syncthreads();
    }

    // --- Epilogue: bias, bf16, stage through LDS, coalesced b128 stores. ---
    // Uniform z-branch hoisted OUTSIDE the loops so inner code is branch-free.
    __bf16* lC = (__bf16*)p1smem;   // reuse (loop ended with __syncthreads)
    if (z == 2) {
#pragma unroll
        for (int mt = 0; mt < 2; ++mt)
#pragma unroll
            for (int nt = 0; nt < 4; ++nt) {
                const int colL = wn + nt * 16 + (lane & 15);
                const float bv_ = bias[n0 + colL];
#pragma unroll
                for (int r = 0; r < 8; ++r) {
                    const int rowL = wm + mt * 16 + r + ((lane >> 4) << 3);
                    lC[colL * CSTR + rowL] = (__bf16)(acc[mt][nt][r] + bv_);
                }
            }
    } else {
#pragma unroll
        for (int mt = 0; mt < 2; ++mt)
#pragma unroll
            for (int nt = 0; nt < 4; ++nt) {
                const int colL = wn + nt * 16 + (lane & 15);
                const float bv_ = bias[n0 + colL];
#pragma unroll
                for (int r = 0; r < 8; ++r) {
                    const int rowL = wm + mt * 16 + r + ((lane >> 4) << 3);
                    lC[rowL * CSTR + colL] = (__bf16)(acc[mt][nt][r] + bv_);
                }
            }
    }
    __syncthreads();

    if (z != 2) {
        __bf16* dst = (z == 0) ? Qb : Kb;
#pragma unroll
        for (int i = 0; i < 8; ++i) {
            const int idx = tid + i * 256;     // 128 rows x 16 chunks
            const int row = idx >> 4;
            const int ch  = idx & 15;
            *(uint4*)(dst + (size_t)(m0 + row) * D + n0 + ch * 8) =
                *(const uint4*)(lC + row * CSTR + ch * 8);
        }
    } else {
        const int bb = m0 >> 11;         // / S
        const int s0 = m0 & (S - 1);     // % S
#pragma unroll
        for (int i = 0; i < 8; ++i) {
            const int idx = tid + i * 256;     // 128 cols x 16 s-chunks
            const int col = idx >> 4;
            const int rc  = idx & 15;
            *(uint4*)(Vt + ((size_t)bb * D + n0 + col) * S + s0 + rc * 8) =
                *(const uint4*)(lC + col * CSTR + rc * 8);
        }
    }
}

// ---------------------------------------------------------------------------
// Phase 2: flash attention. Block = (batch, 32 queries), 8 waves.
// Wave w: scores for its 16-key slice of each 128-key block, then PV for its
// 128-wide output-dim slice. Q staged in LDS (async copy); K/V fragments
// straight from global (contiguous 16B per lane). Next-K-block prefetch.
// ---------------------------------------------------------------------------
#define QSTR 1032   // 1024 + 8 pad (2064B rows: 16B aligned, conflict-free)
#define SSTR 132
#define PSTR 136    // 128 + 8 pad (272B rows)
#define FA_SMEM (32 * QSTR * 2 + 32 * SSTR * 4 + 32 * PSTR * 2 + 3 * 32 * 4)

__global__ __launch_bounds__(256) void flash_attn_kernel(
    const __bf16* __restrict__ Qb, const __bf16* __restrict__ Kb,
    const __bf16* __restrict__ Vt, float* __restrict__ Out)
{
    constexpr int D = 1024, S = 2048;
    extern __shared__ char smem[];
    __bf16* qlds = (__bf16*)smem;                                    // 66048 B
    float*  slds = (float*)(smem + 32 * QSTR * 2);                   // 16896 B
    __bf16* plds = (__bf16*)(smem + 32 * QSTR * 2 + 32 * SSTR * 4);  //  8704 B
    float*  mrow = (float*)(smem + 32 * QSTR * 2 + 32 * SSTR * 4 + 32 * PSTR * 2);
    float*  lrow = mrow + 32;
    float*  srow = lrow + 32;

    const int tid  = threadIdx.x;
    const int lane = tid & 31;
    const int w    = tid >> 5;
    const int b    = blockIdx.y;
    const int q0   = blockIdx.x * 32;
    const int n_off = w * 128;

    const __bf16* Qbase = Qb + ((size_t)b * S + q0) * D;
    const __bf16* Kbase = Kb + (size_t)b * S * D;
    const __bf16* Vbase = Vt + (size_t)b * D * S;

    // Stage 32x1024 Q tile into LDS (async global->LDS, ASYNCcnt-tracked).
#pragma unroll
    for (int i = 0; i < 16; ++i) {
        const int idx = tid + i * 256;
        const int row = idx >> 7;
        const int ch  = idx & 127;
#if HAVE_ASYNC_LDS
        __builtin_amdgcn_global_load_async_to_lds_b128(
            (__attribute__((address_space(1))) v4i_*)(Qbase + (size_t)row * D + ch * 8),
            (__attribute__((address_space(3))) v4i_*)(qlds + row * QSTR + ch * 8),
            0, 0);
#else
        *(uint4*)(qlds + row * QSTR + ch * 8) =
            *(const uint4*)(Qbase + (size_t)row * D + ch * 8);
#endif
    }
#if HAVE_ASYNC_LDS
    __builtin_amdgcn_s_wait_asynccnt(0);
#endif
    if (tid < 32) { mrow[tid] = -3.4e38f; lrow[tid] = 0.f; }

    v8f acc[2][8];
#pragma unroll
    for (int mt = 0; mt < 2; ++mt)
#pragma unroll
        for (int nt = 0; nt < 8; ++nt) acc[mt][nt] = vzero8f();

    __syncthreads();

    const float sscale = 0.03125f;  // 1/sqrt(1024)
    const int srrow  = tid >> 3;        // softmax: 8 threads per row
    const int scol0  = (tid & 7) * 16;

    for (int kb = 0; kb < S / 128; ++kb) {
        // ---- scores: this wave's 16 keys x all 32 queries, contraction D ----
        v8f sacc[2];
        sacc[0] = vzero8f(); sacc[1] = vzero8f();
        const int key = kb * 128 + w * 16 + (lane & 15);
        const __bf16* kptr = Kbase + (size_t)key * D;
        // L2 prefetch of next key block's row for this lane.
        if (kb + 1 < S / 128)
            __builtin_prefetch(kptr + (size_t)128 * D, 0, 1);
#pragma unroll 4
        for (int ks = 0; ks < D / 32; ++ks) {
            const v16bf bfrag = fragB_ld(kptr + ks * 32, lane);
#pragma unroll
            for (int mt = 0; mt < 2; ++mt) {
                const v16bf afrag =
                    fragA_ld(qlds + (mt * 16 + (lane & 15)) * QSTR + ks * 32, lane);
                sacc[mt] = WMMA_BF16(afrag, bfrag, sacc[mt]);
            }
        }
        {
            const int col = w * 16 + (lane & 15);
#pragma unroll
            for (int mt = 0; mt < 2; ++mt)
#pragma unroll
                for (int r = 0; r < 8; ++r) {
                    const int row = mt * 16 + r + ((lane >> 4) << 3);
                    slds[row * SSTR + col] = sacc[mt][r] * sscale;
                }
        }
        __syncthreads();

        // ---- online softmax: 8 threads/row, shfl_xor reduce in 8-lane grp ----
        {
            const float m_old = mrow[srrow];
            float mx = m_old;
#pragma unroll
            for (int c = 0; c < 16; ++c)
                mx = fmaxf(mx, slds[srrow * SSTR + scol0 + c]);
            mx = fmaxf(mx, __shfl_xor(mx, 1, 32));
            mx = fmaxf(mx, __shfl_xor(mx, 2, 32));
            mx = fmaxf(mx, __shfl_xor(mx, 4, 32));
            float sum = 0.f;
#pragma unroll
            for (int c = 0; c < 16; ++c) {
                const float e = __expf(slds[srrow * SSTR + scol0 + c] - mx);
                sum += e;
                plds[srrow * PSTR + scol0 + c] = (__bf16)e;
            }
            sum += __shfl_xor(sum, 1, 32);
            sum += __shfl_xor(sum, 2, 32);
            sum += __shfl_xor(sum, 4, 32);
            if ((tid & 7) == 0) {
                const float sc = __expf(m_old - mx);
                mrow[srrow] = mx;
                lrow[srrow] = lrow[srrow] * sc + sum;
                srow[srrow] = sc;
            }
        }
        __syncthreads();

        // ---- rescale accumulators ----
#pragma unroll
        for (int mt = 0; mt < 2; ++mt)
#pragma unroll
            for (int r = 0; r < 8; ++r) {
                const int row = mt * 16 + r + ((lane >> 4) << 3);
                const float sc = srow[row];
#pragma unroll
                for (int nt = 0; nt < 8; ++nt) acc[mt][nt][r] *= sc;
            }

        // ---- PV: contraction over 128 keys, wave's 128-dim output slice ----
#pragma unroll
        for (int ks = 0; ks < 4; ++ks) {
            v16bf afrag[2];
#pragma unroll
            for (int mt = 0; mt < 2; ++mt)
                afrag[mt] = fragA_ld(plds + (mt * 16 + (lane & 15)) * PSTR + ks * 32, lane);
#pragma unroll
            for (int nt = 0; nt < 8; ++nt) {
                const int n = n_off + nt * 16 + (lane & 15);
                const v16bf bfrag =
                    fragB_ld(Vbase + (size_t)n * S + kb * 128 + ks * 32, lane);
#pragma unroll
                for (int mt = 0; mt < 2; ++mt)
                    acc[mt][nt] = WMMA_BF16(afrag[mt], bfrag, acc[mt][nt]);
            }
        }
        // Next iteration's barriers protect slds/plds/srow reuse.
    }

    // ---- epilogue: normalize by row sum, f32 output ----
#pragma unroll
    for (int mt = 0; mt < 2; ++mt)
#pragma unroll
        for (int r = 0; r < 8; ++r) {
            const int row = mt * 16 + r + ((lane >> 4) << 3);
            const float inv = 1.0f / lrow[row];
            float* orow = Out + ((size_t)b * S + q0 + row) * D;
#pragma unroll
            for (int nt = 0; nt < 8; ++nt) {
                const int n = n_off + nt * 16 + (lane & 15);
                orow[n] = acc[mt][nt][r] * inv;
            }
        }
}

// ---------------------------------------------------------------------------
extern "C" void kernel_launch(void* const* d_in, const int* in_sizes, int n_in,
                              void* d_out, int out_size, void* d_ws, size_t ws_size,
                              hipStream_t stream) {
    constexpr int B = 4, S = 2048, D = 1024;
    const float* X  = (const float*)d_in[0];
    const float* Wq = (const float*)d_in[1];
    const float* bq = (const float*)d_in[2];
    const float* Wk = (const float*)d_in[3];
    const float* bk = (const float*)d_in[4];
    const float* Wv = (const float*)d_in[5];
    const float* bv = (const float*)d_in[6];

    char* ws = (char*)d_ws;
    const size_t SZ = (size_t)B * S * D * sizeof(__bf16);  // 16 MB each
    __bf16* Qb = (__bf16*)(ws);
    __bf16* Kb = (__bf16*)(ws + SZ);
    __bf16* Vt = (__bf16*)(ws + 2 * SZ);   // V transposed [B, D, S]

    qkv_gemm_kernel<<<dim3((B * S) / 128, D / 128, 3), 256, 0, stream>>>(
        X, Wq, Wk, Wv, bq, bk, bv, Qb, Kb, Vt);

    flash_attn_kernel<<<dim3(S / 32, B), 256, FA_SMEM, stream>>>(
        Qb, Kb, Vt, (float*)d_out);
}
